// BlockMask_58557584114275
// MI455X (gfx1250) — matrix operand order repaired and checked by reading, not compiled, not run on hardware
//
#include <hip/hip_runtime.h>
#include <hip/hip_bf16.h>

// ---------------------------------------------------------------------------
// Multi-branch transformer block for MI455X (gfx1250).
// bf16 WMMA GEMMs with double-buffered ASYNC global->LDS tile DMA.
// ---------------------------------------------------------------------------

typedef __attribute__((ext_vector_type(16))) __bf16 v16bf;
typedef __attribute__((ext_vector_type(8)))  float  v8f;

constexpr int kB   = 32;
constexpr int kN   = 129;
constexpr int kC   = 768;
constexpr int kH   = 12;
constexpr int kHID = 3072;
constexpr int kN3  = 3 * kN;     // 387
constexpr float kEPS = 1e-5f;
constexpr float kNEG = -65504.0f;

__device__ __forceinline__ v8f wmma_bf16(v16bf a, v16bf b, v8f c) {
  return __builtin_amdgcn_wmma_f32_16x16x32_bf16(false, a, false, b, (short)0, c,
                                                 false, false);
}

// A-fragment (16x32 bf16): rowp points at k-window start for this lane's row.
// halves 0..15 -> k = g*16 + h (contiguous)
__device__ __forceinline__ v16bf frag_a32(const __bf16* rowp, int g) {
  v16bf a;
  ((uint4*)&a)[0] = *(const uint4*)(rowp + g * 16);
  ((uint4*)&a)[1] = *(const uint4*)(rowp + g * 16 + 8);
  return a;
}

// B-fragment (32x16 bf16): p points at 16 contiguous halves (k = g*16 .. +15)
// for this lane's column.
__device__ __forceinline__ v16bf frag_b16c(const __bf16* p) {
  v16bf b;
  ((uint4*)&b)[0] = *(const uint4*)(p);
  ((uint4*)&b)[1] = *(const uint4*)(p + 8);
  return b;
}

// 16-byte async DMA global -> LDS (GLOBAL_LOAD_ASYNC_TO_LDS_B128, ASYNCcnt).
__device__ __forceinline__ void async_cp16(const void* gptr, void* lptr) {
  unsigned lp = (unsigned)(unsigned long long)lptr;   // wave-relative LDS addr
  asm volatile("global_load_async_to_lds_b128 %0, %1, off"
               :: "v"(lp), "v"(gptr) : "memory");
}

__device__ __forceinline__ void wait_async0() {
#if __has_builtin(__builtin_amdgcn_s_wait_asynccnt)
  __builtin_amdgcn_s_wait_asynccnt(0);
#else
  asm volatile("s_wait_asynccnt 0" ::: "memory");
#endif
}

// ---------------------------------------------------------------------------
// fp32 -> bf16 weight conversion (dense)
__global__ __launch_bounds__(256) void k_cvt_bf16(const float* __restrict__ src,
                                                  __bf16* __restrict__ dst, int n) {
  int i = (blockIdx.x * 256 + threadIdx.x) * 4;
  if (i < n) {
    float4 v = *(const float4*)(src + i);
    dst[i + 0] = (__bf16)v.x;
    dst[i + 1] = (__bf16)v.y;
    dst[i + 2] = (__bf16)v.z;
    dst[i + 3] = (__bf16)v.w;
  }
}

// fp32 [4][K][Nc] -> bf16 transposed [4][Nc][K] (so GEMM B tiles are
// contiguous per output column and can be DMA'd without transpose)
__global__ __launch_bounds__(256) void k_cvt_t(const float* __restrict__ src,
                                               __bf16* __restrict__ dst,
                                               int K, int Nc) {
  long i = (long)blockIdx.x * 256 + threadIdx.x;
  long per = (long)K * Nc;
  if (i >= 4 * per) return;
  int b = (int)(i / per);
  long rem = i % per;
  int n = (int)(rem / K), k = (int)(rem % K);
  dst[i] = (__bf16)src[(long)b * per + (long)k * Nc + n];
}

// m3[b, s] with s in [0,3N): tiled mask, leading 1 per branch segment
__global__ __launch_bounds__(256) void k_mask(const float* __restrict__ mk,
                                              float* __restrict__ m3) {
  int i = blockIdx.x * 256 + threadIdx.x;
  if (i >= kB * kN3) return;
  int b = i / kN3, s = i % kN3, n = s % kN;
  m3[i] = (n == 0) ? 1.0f : mk[b * (kN - 1) + (n - 1)];
}

// scatter RGB/NIR/TIR into x4[b, br*N + n, c]  (free concat)
__global__ __launch_bounds__(256) void k_init_x4(const float* __restrict__ R,
                                                 const float* __restrict__ Nr,
                                                 const float* __restrict__ T,
                                                 float* __restrict__ x4) {
  long i = (long)(blockIdx.x * 256 + threadIdx.x) * 4;
  const long total = (long)kB * kN3 * kC;
  if (i >= total) return;
  long row = i / kC;
  int c = (int)(i % kC);
  int b = (int)(row / kN3), s = (int)(row % kN3);
  int br = s / kN, n = s % kN;
  const float* src = (br == 0) ? R : ((br == 1) ? Nr : T);
  float4 v = *(const float4*)(src + ((long)(b * kN + n) * kC + c));
  *(float4*)(x4 + i) = v;
}

// ---------------------------------------------------------------------------
// LayerNorm * mask. Reads x with batch mapping (row = b*batchStride + n*C).
// outF != null  -> write fp32 dense (final LN into d_out)
// else          -> write bf16 dense [M, C]
__global__ __launch_bounds__(256) void k_ln(const float* __restrict__ x,
                                            int batchStride, int Ntok,
                                            const float* __restrict__ gam,
                                            const float* __restrict__ bet,
                                            const float* __restrict__ m3, int mRowOff,
                                            __bf16* __restrict__ outB,
                                            float* __restrict__ outF) {
  __shared__ float red[256];
  __shared__ float s_mu, s_rstd;
  int t = blockIdx.x, b = t / Ntok, n = t % Ntok;
  const float* row = x + (size_t)b * batchStride + (size_t)n * kC;
  int tid = threadIdx.x;
  float v0 = row[tid], v1 = row[tid + 256], v2 = row[tid + 512];
  red[tid] = v0 + v1 + v2;
  __syncthreads();
  for (int o = 128; o; o >>= 1) { if (tid < o) red[tid] += red[tid + o]; __syncthreads(); }
  if (tid == 0) s_mu = red[0] * (1.0f / kC);
  __syncthreads();
  float mu = s_mu;
  float d0 = v0 - mu, d1 = v1 - mu, d2 = v2 - mu;
  red[tid] = d0 * d0 + d1 * d1 + d2 * d2;
  __syncthreads();
  for (int o = 128; o; o >>= 1) { if (tid < o) red[tid] += red[tid + o]; __syncthreads(); }
  if (tid == 0) s_rstd = rsqrtf(red[0] * (1.0f / kC) + kEPS);
  __syncthreads();
  float rstd = s_rstd;
  float mval = m3[b * kN3 + mRowOff + n];
  float y0 = (d0 * rstd * gam[tid] + bet[tid]) * mval;
  float y1 = (d1 * rstd * gam[tid + 256] + bet[tid + 256]) * mval;
  float y2 = (d2 * rstd * gam[tid + 512] + bet[tid + 512]) * mval;
  size_t o0 = (size_t)t * kC;
  if (outF) {
    outF[o0 + tid] = y0; outF[o0 + tid + 256] = y1; outF[o0 + tid + 512] = y2;
  } else {
    outB[o0 + tid] = (__bf16)y0;
    outB[o0 + tid + 256] = (__bf16)y1;
    outB[o0 + tid + 512] = (__bf16)y2;
  }
}

// ---------------------------------------------------------------------------
// bf16 WMMA GEMM: D[M,Nc] = A[M,K] * W^T'[Nc,K]  (Wt is pre-transposed)
// Block tile 128x128, 8 waves (4m x 2n) of 32x64 wave tiles, K step 32.
// Double-buffered LDS; tiles staged with GLOBAL_LOAD_ASYNC_TO_LDS_B128.
// epi: 0 = store bf16 dense; 1 = GELU -> bf16 dense;
//      2 = residual add into fp32 outF with batch row mapping.
#define LDK 40   // padded LDS row stride in halves (80B, 16B-aligned, no 4-way bank dup)
__global__ __launch_bounds__(256) void k_gemm(const __bf16* __restrict__ A, int M,
                                              int K, const __bf16* __restrict__ Wt,
                                              int Nc, int epi, int Ntok,
                                              int outBatchStride,
                                              __bf16* __restrict__ outB,
                                              float* __restrict__ outF) {
  __shared__ __align__(16) __bf16 Ald[2][128][LDK];   // [m][k]
  __shared__ __align__(16) __bf16 Btl[2][128][LDK];   // [n][k] (already transposed)
  int tid = threadIdx.x;
  int wave = tid >> 5, lane = tid & 31;
  int g = lane >> 4, lr = lane & 15;
  int wm = wave >> 1, wn = wave & 1;
  int m0 = blockIdx.x * 128, n0 = blockIdx.y * 128;

  // per-thread DMA chunk coords: 512 chunks of 16B per tile, 2 per thread
  int r0 = tid >> 1;                 // unused; the lambdas use row = idx >> 2
  int r1 = (tid + 256) >> 1;         // hmm-free: recompute inside lambda instead

  auto issueA = [&](int k0, int buf) {
#pragma unroll
    for (int it = 0; it < 2; it++) {
      int idx = tid + it * 256;            // 0..511
      int r = idx >> 2, c8 = idx & 3;      // 4 x 16B chunks per 64B row
      int gr = m0 + r; if (gr >= M) gr = M - 1;   // clamp: rows >= M are junk but never stored
      async_cp16(A + (size_t)gr * K + k0 + c8 * 8, &Ald[buf][r][c8 * 8]);
    }
  };
  auto issueB = [&](int k0, int buf) {
#pragma unroll
    for (int it = 0; it < 2; it++) {
      int idx = tid + it * 256;
      int r = idx >> 2, c8 = idx & 3;
      async_cp16(Wt + (size_t)(n0 + r) * K + k0 + c8 * 8, &Btl[buf][r][c8 * 8]);
    }
  };

  v8f zero = {0.f, 0.f, 0.f, 0.f, 0.f, 0.f, 0.f, 0.f};
  v8f c[2][4];
#pragma unroll
  for (int mi = 0; mi < 2; mi++)
#pragma unroll
    for (int ni = 0; ni < 4; ni++) c[mi][ni] = zero;

  issueA(0, 0);
  issueB(0, 0);
  wait_async0();
  __syncthreads();

  int nk = K >> 5;
  for (int kt = 0; kt < nk; kt++) {
    int buf = kt & 1;
    if (kt + 1 < nk) {                 // DMA next tile while computing this one
      issueA((kt + 1) << 5, buf ^ 1);
      issueB((kt + 1) << 5, buf ^ 1);
    }
    v16bf bf[4];
#pragma unroll
    for (int ni = 0; ni < 4; ni++)
      bf[ni] = frag_b16c(&Btl[buf][wn * 64 + ni * 16 + lr][g * 16]);
#pragma unroll
    for (int mi = 0; mi < 2; mi++) {
      v16bf a = frag_a32(&Ald[buf][wm * 32 + mi * 16 + lr][0], g);
#pragma unroll
      for (int ni = 0; ni < 4; ni++) c[mi][ni] = wmma_bf16(a, bf[ni], c[mi][ni]);
    }
    wait_async0();                     // next-tile DMA landed
    __syncthreads();                   // all waves done reading buf
  }

  // epilogue: C/D layout => row = v + g*8, col = lr within each 16x16 tile
#pragma unroll
  for (int mi = 0; mi < 2; mi++)
#pragma unroll
    for (int ni = 0; ni < 4; ni++)
#pragma unroll
      for (int v = 0; v < 8; v++) {
        int t = m0 + wm * 32 + mi * 16 + g * 8 + v;
        if (t >= M) continue;
        int n = n0 + wn * 64 + ni * 16 + lr;
        float x = c[mi][ni][v];
        if (epi == 0) {
          outB[(size_t)t * Nc + n] = (__bf16)x;
        } else if (epi == 1) {
          float ge = 0.5f * x * (1.0f + erff(x * 0.70710678118654752f));
          outB[(size_t)t * Nc + n] = (__bf16)ge;
        } else {
          size_t off = (size_t)(t / Ntok) * outBatchStride +
                       (size_t)(t % Ntok) * kC + n;
          outF[off] += x;   // residual add
        }
      }
  (void)r0; (void)r1;
}

// ---------------------------------------------------------------------------
// Attention per (b, head, 32-query tile): S = Q K^T / 8, masked softmax, P V.
// qkv: bf16 [B*Ntok, 3C] with col = which*C + head*64 + d.
// Dynamic LDS: Sf fp32 [32][NpadC] then Pb bf16 [32][NpadC].
__global__ __launch_bounds__(256) void k_attn(const __bf16* __restrict__ qkv,
                                              const float* __restrict__ m3,
                                              int mRowOff,
                                              __bf16* __restrict__ outp,
                                              int Ntok, int NpadC) {
  extern __shared__ __align__(16) char smem[];
  float* Sf = (float*)smem;
  __bf16* Pb = (__bf16*)(smem + (size_t)32 * NpadC * 4);

  int q0 = blockIdx.x * 32;
  int head = blockIdx.y, b = blockIdx.z;
  int tid = threadIdx.x, wave = tid >> 5, lane = tid & 31;
  int g = lane >> 4, lr = lane & 15;
  const int C3 = 3 * kC;
  const __bf16* qb = qkv + (size_t)b * Ntok * C3;
  const float* mrow = m3 + (size_t)b * kN3 + mRowOff;
  int hd0 = head * 64;
  v8f zero = {0.f, 0.f, 0.f, 0.f, 0.f, 0.f, 0.f, 0.f};

  // Q fragments (2 q-subtiles x 2 k-steps of 32 over hd=64)
  v16bf aq[2][2];
#pragma unroll
  for (int qi = 0; qi < 2; qi++) {
    int n = q0 + qi * 16 + lr; if (n >= Ntok) n = Ntok - 1;
    const __bf16* rp = qb + (size_t)n * C3 + hd0;
    aq[qi][0] = frag_a32(rp, g);
    aq[qi][1] = frag_a32(rp + 32, g);
  }

  // Phase 1: scores into LDS (masked, scaled)
  int Npad16 = NpadC >> 4;
  for (int j = wave; j < Npad16; j += 8) {
    int kk = j * 16 + lr;
    int kkc = (kk < Ntok) ? kk : Ntok - 1;
    const __bf16* kp = qb + (size_t)kkc * C3 + kC + hd0 + g * 16;
    v16bf bk0 = frag_b16c(kp);        // d = g*16+h
    v16bf bk1 = frag_b16c(kp + 32);   // d = 32+g*16+h
    float mk = (kk < Ntok) ? mrow[kk] : 0.0f;
#pragma unroll
    for (int qi = 0; qi < 2; qi++) {
      v8f acc = zero;
      acc = wmma_bf16(aq[qi][0], bk0, acc);
      acc = wmma_bf16(aq[qi][1], bk1, acc);
#pragma unroll
      for (int v = 0; v < 8; v++) {
        int qrow = qi * 16 + g * 8 + v;
        float val = (mk != 0.0f) ? acc[v] * 0.125f : kNEG;
        Sf[qrow * NpadC + kk] = val;
      }
    }
  }
  __syncthreads();

  // Phase 2: softmax per row (8 threads/row), then * m_q, to bf16
  {
    int r = tid >> 3, sl = tid & 7;
    int qq = q0 + r;
    float mq = (qq < Ntok) ? mrow[qq] : 0.0f;
    float* srow = Sf + r * NpadC;
    float mx = -3.4e38f;
    for (int c2 = sl; c2 < NpadC; c2 += 8) mx = fmaxf(mx, srow[c2]);
#pragma unroll
    for (int o = 4; o; o >>= 1) mx = fmaxf(mx, __shfl_xor(mx, o, 8));
    float sum = 0.0f;
    for (int c2 = sl; c2 < NpadC; c2 += 8) {
      float e = __expf(srow[c2] - mx);
      srow[c2] = e;
      sum += e;
    }
#pragma unroll
    for (int o = 4; o; o >>= 1) sum += __shfl_xor(sum, o, 8);
    float inv = mq / sum;
    __bf16* prow = Pb + r * NpadC;
    for (int c2 = sl; c2 < NpadC; c2 += 8) prow[c2] = (__bf16)(srow[c2] * inv);
  }
  __syncthreads();

  // Phase 3: O = P V. 8 output 16x16 tiles -> one per wave.
  {
    int qi = wave >> 2, di = wave & 3;
    v8f acc = zero;
    for (int k0 = 0; k0 < NpadC; k0 += 32) {
      v16bf ap = frag_a32(Pb + (size_t)(qi * 16 + lr) * NpadC + k0, g);
      v16bf bv;
#pragma unroll
      for (int hh = 0; hh < 16; hh++) {
        int kk = k0 + g * 16 + hh;
        int kkc = (kk < Ntok) ? kk : Ntok - 1;
        __bf16 x = qb[(size_t)kkc * C3 + 2 * kC + hd0 + di * 16 + lr];
        bv[hh] = (kk < Ntok) ? x : (__bf16)0.0f;
      }
      acc = wmma_bf16(ap, bv, acc);
    }
#pragma unroll
    for (int v = 0; v < 8; v++) {
      int qq = q0 + qi * 16 + g * 8 + v;
      if (qq < Ntok)
        outp[((size_t)b * Ntok + qq) * kC + hd0 + di * 16 + lr] = (__bf16)acc[v];
    }
  }
}

// ---------------------------------------------------------------------------
extern "C" void kernel_launch(void* const* d_in, const int* in_sizes, int n_in,
                              void* d_out, int out_size, void* d_ws, size_t ws_size,
                              hipStream_t stream) {
  const float* RGB   = (const float*)d_in[0];
  const float* NIR   = (const float*)d_in[1];
  const float* TIR   = (const float*)d_in[2];
  const float* maskp = (const float*)d_in[3];
  // d_in[4] = label (unused)
  const float* ln_g1 = (const float*)d_in[5];
  const float* ln_b1 = (const float*)d_in[6];
  const float* w_qkv = (const float*)d_in[7];
  const float* w_proj= (const float*)d_in[8];
  const float* ln_g2 = (const float*)d_in[9];
  const float* ln_b2 = (const float*)d_in[10];
  const float* w_fc1 = (const float*)d_in[11];
  const float* w_fc2 = (const float*)d_in[12];
  const float* out_g = (const float*)d_in[13];
  const float* out_b = (const float*)d_in[14];

  char* ws = (char*)d_ws;
  size_t off = 0;
  auto alloc = [&](size_t bytes) -> char* {
    char* p = ws + off;
    off += (bytes + 255) & ~(size_t)255;
    return p;
  };
  // transposed bf16 weights: [4][Nc][K]
  __bf16* WQ  = (__bf16*)alloc((size_t)4 * kC * 3 * kC * 2);
  __bf16* WP  = (__bf16*)alloc((size_t)4 * kC * kC * 2);
  __bf16* W1  = (__bf16*)alloc((size_t)4 * kC * kHID * 2);
  __bf16* W2  = (__bf16*)alloc((size_t)4 * kHID * kC * 2);
  float*  m3  = (float*)alloc((size_t)kB * kN3 * 4);
  float*  x4  = (float*)alloc((size_t)kB * kN3 * kC * 4);
  __bf16* xmA = (__bf16*)alloc((size_t)kB * kN3 * kC * 2);   // xm / attn_out
  __bf16* big = (__bf16*)alloc((size_t)kB * kN3 * kHID * 2); // qkv then h1

  auto cvtT = [&](const float* s, __bf16* d, int K, int Nc) {
    long tot = (long)4 * K * Nc;
    k_cvt_t<<<dim3((unsigned)((tot + 255) / 256)), 256, 0, stream>>>(s, d, K, Nc);
  };
  cvtT(w_qkv,  WQ, kC, 3 * kC);
  cvtT(w_proj, WP, kC, kC);
  cvtT(w_fc1,  W1, kC, kHID);
  cvtT(w_fc2,  W2, kHID, kC);

  k_mask<<<dim3((kB * kN3 + 255) / 256), 256, 0, stream>>>(maskp, m3);
  k_init_x4<<<dim3((unsigned)(((size_t)kB * kN3 * kC / 4 + 255) / 256)), 256, 0,
              stream>>>(RGB, NIR, TIR, x4);

  auto gemm = [&](const __bf16* A, int M, int K, const __bf16* Wt, int Nc, int epi,
                  int Ntok, __bf16* outB, float* outF) {
    dim3 grid((M + 127) / 128, Nc / 128);
    k_gemm<<<grid, 256, 0, stream>>>(A, M, K, Wt, Nc, epi, Ntok, kN3 * kC, outB, outF);
  };

  auto run_branch = [&](int i, int Ntok, int brOff) {
    int M = kB * Ntok;
    float* xb = x4 + (size_t)brOff * kC;
    // LN1 * mask -> bf16
    k_ln<<<M, 256, 0, stream>>>(xb, kN3 * kC, Ntok, ln_g1 + i * kC, ln_b1 + i * kC,
                                m3, brOff, xmA, (float*)nullptr);
    // QKV
    gemm(xmA, M, kC, WQ + (size_t)i * kC * 3 * kC, 3 * kC, 0, Ntok, big, nullptr);
    // Attention
    int NpadC = ((Ntok + 31) / 32) * 32;
    dim3 ag((Ntok + 31) / 32, kH, kB);
    size_t shb = (size_t)32 * NpadC * 6;   // fp32 scores + bf16 probs
    k_attn<<<ag, 256, shb, stream>>>(big, m3, brOff, xmA, Ntok, NpadC);
    // proj + residual into x4
    gemm(xmA, M, kC, WP + (size_t)i * kC * kC, kC, 2, Ntok, nullptr, xb);
    // LN2 * mask -> bf16
    k_ln<<<M, 256, 0, stream>>>(xb, kN3 * kC, Ntok, ln_g2 + i * kC, ln_b2 + i * kC,
                                m3, brOff, xmA, (float*)nullptr);
    // FC1 + GELU -> bf16
    gemm(xmA, M, kC, W1 + (size_t)i * kC * kHID, kHID, 1, Ntok, big, nullptr);
    // FC2 + residual into x4
    gemm(big, M, kHID, W2 + (size_t)i * kHID * kC, kC, 2, Ntok, nullptr, xb);
  };

  run_branch(0, kN, 0);
  run_branch(1, kN, kN);
  run_branch(2, kN, 2 * kN);
  run_branch(3, kN3, 0);

  // final LN * mask -> fp32 output [B, 3N, C]
  k_ln<<<kB * kN3, 256, 0, stream>>>(x4, kN3 * kC, kN3, out_g, out_b, m3, 0,
                                     (__bf16*)nullptr, (float*)d_out);
}
